// PerformerAttention_50397146251453
// MI455X (gfx1250) — matrix operand order repaired and checked by reading, not compiled, not run on hardware
//
#include <hip/hip_runtime.h>
#include <hip/hip_bf16.h>

#define DM     512
#define HEADS  8
#define DH     64
#define BATCH  2
#define SEQ    2048
#define ROWS   (BATCH * SEQ)   // 4096 tokens
#define BHEADS (BATCH * HEADS) // 16

typedef _Float16 half_t;
typedef _Float16 v16h __attribute__((ext_vector_type(16)));
typedef _Float16 v8h  __attribute__((ext_vector_type(8)));
typedef float    v8f  __attribute__((ext_vector_type(8)));
typedef float    v4f  __attribute__((ext_vector_type(4)));

// ---------------------------------------------------------------------------
// WMMA helpers (wave32, v_wmma_f32_16x16x32_f16). Layouts per CDNA5 ISA 7.12.2
// ---------------------------------------------------------------------------
__device__ __forceinline__ v8f wmma_f16(v16h a, v16h b, v8f c) {
  // (neg_a, A, neg_b, B, c_mod, C, reuse_a, reuse_b)
  return __builtin_amdgcn_wmma_f32_16x16x32_f16(false, a, false, b,
                                                (short)0, c, false, false);
}

__device__ __forceinline__ v8f zero8() {
  v8f z;
#pragma unroll
  for (int i = 0; i < 8; ++i) z[i] = 0.0f;
  return z;
}

// A fragment 16x32 f16, row-major source (lda in elements).
// lane l: row m = l&15 ; K chunks [(l>>4)*8, +8) and +16..+24
__device__ __forceinline__ v16h load_a_f16(const half_t* a, int lda, int lane) {
  const int m = lane & 15;
  const int g = (lane >> 4) * 8;
  const half_t* p = a + (size_t)m * lda + g;
  v8h lo = *(const v8h*)(p);
  v8h hi = *(const v8h*)(p + 16);
  v16h r;
#pragma unroll
  for (int i = 0; i < 8; ++i) { r[i] = lo[i]; r[i + 8] = hi[i]; }
  return r;
}

// A fragment from an f32 row-major source, non-temporal (last-use stream),
// converted to f16 on load.
__device__ __forceinline__ v16h load_a_f32_nt(const float* a, int lda, int lane) {
  const int m = lane & 15;
  const int g = (lane >> 4) * 8;
  const float* p = a + (size_t)m * lda + g;
  v4f x0 = __builtin_nontemporal_load((const v4f*)(p));
  v4f x1 = __builtin_nontemporal_load((const v4f*)(p + 4));
  v4f x2 = __builtin_nontemporal_load((const v4f*)(p + 16));
  v4f x3 = __builtin_nontemporal_load((const v4f*)(p + 20));
  v16h r;
#pragma unroll
  for (int i = 0; i < 4; ++i) {
    r[i]      = (half_t)x0[i];
    r[i + 4]  = (half_t)x1[i];
    r[i + 8]  = (half_t)x2[i];
    r[i + 12] = (half_t)x3[i];
  }
  return r;
}

// B fragment 32x16 f16, row-major K x N source (ldb in elements).
// lane l: col n = l&15 ; K rows [(l>>4)*8, +8) and +16..+24
__device__ __forceinline__ v16h load_b_f16(const half_t* b, int ldb, int lane) {
  const half_t* p = b + (size_t)((lane >> 4) * 8) * ldb + (lane & 15);
  v16h r;
#pragma unroll
  for (int i = 0; i < 8; ++i) { r[i] = p[(size_t)i * ldb]; r[i + 8] = p[(size_t)(i + 16) * ldb]; }
  return r;
}

// ---------------------------------------------------------------------------
// Prep: x (fp32) -> f16 ; W[n][k] (fp32) -> WT[k][n] (f16)
// ---------------------------------------------------------------------------
__global__ void cvt_x_kernel(const float* __restrict__ x, half_t* __restrict__ xh) {
  int idx = blockIdx.x * 256 + threadIdx.x;   // ROWS*DM total
  xh[idx] = (half_t)x[idx];
}

__global__ void wtrans_kernel(const float* __restrict__ W, half_t* __restrict__ WT) {
  int idx = blockIdx.x * 256 + threadIdx.x;   // DM*DM total
  int k = idx >> 9;
  int n = idx & (DM - 1);
  WT[idx] = (half_t)W[(size_t)n * DM + k];
}

// ---------------------------------------------------------------------------
// QKV projection: D = xh @ WT (+bias, relu for q,k).
// Each wave: 64x16 output (4 M-tiles sharing one B fragment per k-step).
// which: 0=Q -> Qh[bh][s][dh], 1=K -> KhT[bh][dh][s] (transposed!), 2=V -> Vh
// ---------------------------------------------------------------------------
__global__ void qkv_kernel(const half_t* __restrict__ xh,
                           const half_t* __restrict__ WqT,
                           const half_t* __restrict__ WkT,
                           const half_t* __restrict__ WvT,
                           const float* __restrict__ bq,
                           const float* __restrict__ bk,
                           const float* __restrict__ bv,
                           half_t* __restrict__ Qh,
                           half_t* __restrict__ KhT,
                           half_t* __restrict__ Vh) {
  const int lane  = threadIdx.x;
  const int which = blockIdx.z;
  const int m0 = blockIdx.x * 64;                       // 4 M-tiles per wave
  const int n0 = (blockIdx.y * 4 + threadIdx.y) * 16;

  const half_t* wt   = (which == 0) ? WqT : (which == 1) ? WkT : WvT;
  const float*  bias = (which == 0) ? bq  : (which == 1) ? bk  : bv;

  v8f acc[4];
#pragma unroll
  for (int t = 0; t < 4; ++t) acc[t] = zero8();

#pragma unroll 2
  for (int kk = 0; kk < DM; kk += 32) {
    v16h b = load_b_f16(wt + (size_t)kk * DM + n0, DM, lane);
#pragma unroll
    for (int t = 0; t < 4; ++t) {
      v16h a = load_a_f16(xh + (size_t)(m0 + 16 * t) * DM + kk, DM, lane);
      acc[t] = wmma_f16(a, b, acc[t]);
    }
  }

  const int n  = n0 + (lane & 15);
  const int hh = n >> 6;        // head
  const int dh = n & (DH - 1);  // dim in head
  const float bv_ = bias[n];
#pragma unroll
  for (int t = 0; t < 4; ++t) {
#pragma unroll
    for (int v = 0; v < 8; ++v) {
      const int m  = m0 + 16 * t + v + 8 * (lane >> 4);  // global token
      const int bb = m >> 11;                            // batch
      const int s  = m & (SEQ - 1);                      // position
      float val = acc[t][v] + bv_;
      const size_t bh = (size_t)bb * HEADS + hh;
      if (which == 0) {
        val = fmaxf(val, 0.0f);
        Qh[(bh * SEQ + s) * DH + dh] = (half_t)val;
      } else if (which == 1) {
        val = fmaxf(val, 0.0f);
        KhT[(bh * DH + dh) * SEQ + s] = (half_t)val;
      } else {
        Vh[(bh * SEQ + s) * DH + dh] = (half_t)val;
      }
    }
  }
}

// ---------------------------------------------------------------------------
// k_sum[bh][s] = sum_f relu(k)[bh][s][f]   (K already relu'd, stored as KhT)
// ---------------------------------------------------------------------------
__global__ void ksum_kernel(const half_t* __restrict__ KhT, float* __restrict__ ksum) {
  int idx = blockIdx.x * 256 + threadIdx.x;   // BHEADS*SEQ
  int bh = idx >> 11;
  int s  = idx & (SEQ - 1);
  float acc = 0.0f;
#pragma unroll 8
  for (int f = 0; f < DH; ++f)
    acc += (float)KhT[((size_t)bh * DH + f) * SEQ + s];
  ksum[idx] = acc;
}

// ---------------------------------------------------------------------------
// scores[bh][n][m] = (Q[n] . K[m]) / (ksum[n] + eps) -> attn output region.
// Each wave: 16x64 output (4 key-tiles sharing one Q fragment per k-step).
// attn stores are non-temporal (256 MB stream, must not thrash L2).
// ---------------------------------------------------------------------------
__global__ void scores_kernel(const half_t* __restrict__ Qh,
                              const half_t* __restrict__ KhT,
                              const float* __restrict__ ksum,
                              float* __restrict__ attn) {
  const int lane = threadIdx.x;
  const int bh = blockIdx.z;
  const int m0 = blockIdx.x * 16;                            // query tile
  const int n0 = (blockIdx.y * 4 + threadIdx.y) * 64;        // 4 key tiles

  const half_t* Q  = Qh  + (size_t)bh * SEQ * DH;
  const half_t* KT = KhT + (size_t)bh * DH * SEQ;

  v8f acc[4];
#pragma unroll
  for (int t = 0; t < 4; ++t) acc[t] = zero8();

#pragma unroll
  for (int kk = 0; kk < DH; kk += 32) {
    v16h a = load_a_f16(Q + (size_t)m0 * DH + kk, DH, lane);
#pragma unroll
    for (int t = 0; t < 4; ++t) {
      v16h b = load_b_f16(KT + (size_t)kk * SEQ + n0 + 16 * t, SEQ, lane);
      acc[t] = wmma_f16(a, b, acc[t]);
    }
  }

#pragma unroll
  for (int v = 0; v < 8; ++v) {
    const int m = m0 + v + 8 * (lane >> 4);   // query index
    const float den = ksum[bh * SEQ + m] + 1e-6f;
    float* row = attn + ((size_t)bh * SEQ + m) * SEQ + (lane & 15);
#pragma unroll
    for (int t = 0; t < 4; ++t)
      __builtin_nontemporal_store(acc[t][v] / den, row + n0 + 16 * t);
  }
}

// ---------------------------------------------------------------------------
// In-place row softmax over 2048 columns. One 256-thread block per row.
// Non-temporal row traffic (single-use stream).
// ---------------------------------------------------------------------------
__global__ void softmax_kernel(float* __restrict__ attn) {
  __shared__ float red[256];
  const int t = threadIdx.x;
  float* p = attn + (size_t)blockIdx.x * SEQ;

  float v[8];
  float m = -3.402823e38f;
#pragma unroll
  for (int j = 0; j < 8; ++j) {
    v[j] = __builtin_nontemporal_load(p + t + j * 256);
    m = fmaxf(m, v[j]);
  }
  red[t] = m;
  __syncthreads();
  for (int s = 128; s > 0; s >>= 1) {
    if (t < s) red[t] = fmaxf(red[t], red[t + s]);
    __syncthreads();
  }
  m = red[0];
  __syncthreads();

  float sum = 0.0f;
#pragma unroll
  for (int j = 0; j < 8; ++j) { v[j] = __expf(v[j] - m); sum += v[j]; }
  red[t] = sum;
  __syncthreads();
  for (int s = 128; s > 0; s >>= 1) {
    if (t < s) red[t] += red[t + s];
    __syncthreads();
  }
  const float inv = 1.0f / red[0];
#pragma unroll
  for (int j = 0; j < 8; ++j)
    __builtin_nontemporal_store(v[j] * inv, p + t + j * 256);
}

// ---------------------------------------------------------------------------
// OutH[b][s][h*64+dh] = attn[bh] @ Vh[bh].
// Each wave: 16x64 output (one attn A fragment drives all 4 dh-tiles; V fully
// register-reused). attn loads are non-temporal last-use.
// ---------------------------------------------------------------------------
__global__ void av_kernel(const float* __restrict__ attn,
                          const half_t* __restrict__ Vh,
                          half_t* __restrict__ OutH) {
  const int lane = threadIdx.x;
  const int bh = blockIdx.z;
  const int m0 = (blockIdx.x * 4 + threadIdx.y) * 16;   // one M-tile per wave

  const float*  P = attn + ((size_t)bh * SEQ + m0) * SEQ;
  const half_t* V = Vh + (size_t)bh * SEQ * DH;

  v8f acc[4];
#pragma unroll
  for (int t = 0; t < 4; ++t) acc[t] = zero8();

#pragma unroll 2
  for (int kk = 0; kk < SEQ; kk += 32) {
    v16h a = load_a_f32_nt(P + kk, SEQ, lane);
#pragma unroll
    for (int t = 0; t < 4; ++t) {
      v16h b = load_b_f16(V + (size_t)kk * DH + 16 * t, DH, lane);
      acc[t] = wmma_f16(a, b, acc[t]);
    }
  }

  const int bb = bh >> 3;
  const int hh = bh & (HEADS - 1);
#pragma unroll
  for (int t = 0; t < 4; ++t) {
    const int dh = 16 * t + (lane & 15);
#pragma unroll
    for (int v = 0; v < 8; ++v) {
      const int s = m0 + v + 8 * (lane >> 4);
      OutH[((size_t)bb * SEQ + s) * DM + hh * DH + dh] = (half_t)acc[t][v];
    }
  }
}

// ---------------------------------------------------------------------------
// out = OutH @ WoT + bo  (fp32 result).
// Each wave: 64x16 output (4 M-tiles sharing one B fragment per k-step).
// ---------------------------------------------------------------------------
__global__ void proj_kernel(const half_t* __restrict__ OutH,
                            const half_t* __restrict__ WoT,
                            const float* __restrict__ bo,
                            float* __restrict__ out) {
  const int lane = threadIdx.x;
  const int m0 = blockIdx.x * 64;
  const int n0 = (blockIdx.y * 4 + threadIdx.y) * 16;

  v8f acc[4];
#pragma unroll
  for (int t = 0; t < 4; ++t) acc[t] = zero8();

#pragma unroll 2
  for (int kk = 0; kk < DM; kk += 32) {
    v16h b = load_b_f16(WoT + (size_t)kk * DM + n0, DM, lane);
#pragma unroll
    for (int t = 0; t < 4; ++t) {
      v16h a = load_a_f16(OutH + (size_t)(m0 + 16 * t) * DM + kk, DM, lane);
      acc[t] = wmma_f16(a, b, acc[t]);
    }
  }

  const int n = n0 + (lane & 15);
  const float bias = bo[n];
#pragma unroll
  for (int t = 0; t < 4; ++t) {
#pragma unroll
    for (int v = 0; v < 8; ++v) {
      const int m = m0 + 16 * t + v + 8 * (lane >> 4);
      out[(size_t)m * DM + n] = acc[t][v] + bias;
    }
  }
}

// ---------------------------------------------------------------------------
extern "C" void kernel_launch(void* const* d_in, const int* in_sizes, int n_in,
                              void* d_out, int out_size, void* d_ws, size_t ws_size,
                              hipStream_t stream) {
  (void)in_sizes; (void)n_in; (void)out_size; (void)ws_size;
  const float* x  = (const float*)d_in[0];
  const float* Wq = (const float*)d_in[1];
  const float* bq = (const float*)d_in[2];
  const float* Wk = (const float*)d_in[3];
  const float* bk = (const float*)d_in[4];
  const float* Wv = (const float*)d_in[5];
  const float* bv = (const float*)d_in[6];
  const float* Wo = (const float*)d_in[7];
  const float* bo = (const float*)d_in[8];

  float* out  = (float*)d_out;
  float* attn = out + (size_t)ROWS * DM;   // second output region

  char* ws = (char*)d_ws;
  size_t o = 0;
  half_t* xh   = (half_t*)(ws + o); o += (size_t)ROWS * DM * 2;
  half_t* WqT  = (half_t*)(ws + o); o += (size_t)DM * DM * 2;
  half_t* WkT  = (half_t*)(ws + o); o += (size_t)DM * DM * 2;
  half_t* WvT  = (half_t*)(ws + o); o += (size_t)DM * DM * 2;
  half_t* WoT  = (half_t*)(ws + o); o += (size_t)DM * DM * 2;
  half_t* Qh   = (half_t*)(ws + o); o += (size_t)BHEADS * SEQ * DH * 2;
  half_t* KhT  = (half_t*)(ws + o); o += (size_t)BHEADS * DH * SEQ * 2;
  half_t* Vh   = (half_t*)(ws + o); o += (size_t)BHEADS * SEQ * DH * 2;
  half_t* OutH = (half_t*)(ws + o); o += (size_t)ROWS * DM * 2;
  float*  ksum = (float*)(ws + o);  o += (size_t)BHEADS * SEQ * 4;

  const dim3 wblk(32, 4);  // 4 wave32s per block

  // 1) precision prep
  cvt_x_kernel<<<(ROWS * DM) / 256, 256, 0, stream>>>(x, xh);
  wtrans_kernel<<<(DM * DM) / 256, 256, 0, stream>>>(Wq, WqT);
  wtrans_kernel<<<(DM * DM) / 256, 256, 0, stream>>>(Wk, WkT);
  wtrans_kernel<<<(DM * DM) / 256, 256, 0, stream>>>(Wv, WvT);
  wtrans_kernel<<<(DM * DM) / 256, 256, 0, stream>>>(Wo, WoT);

  // 2) QKV projections (WMMA, 64x16 per wave)
  qkv_kernel<<<dim3(ROWS / 64, (DM / 16) / 4, 3), wblk, 0, stream>>>(
      xh, WqT, WkT, WvT, bq, bk, bv, Qh, KhT, Vh);

  // 3) per-position K sum
  ksum_kernel<<<(BHEADS * SEQ) / 256, 256, 0, stream>>>(KhT, ksum);

  // 4) scores = QK^T / (ksum + eps)  (WMMA, 16x64 per wave, NT stores)
  scores_kernel<<<dim3(SEQ / 16, (SEQ / 64) / 4, BHEADS), wblk, 0, stream>>>(
      Qh, KhT, ksum, attn);

  // 5) softmax rows (in place, this is the second kernel output)
  softmax_kernel<<<BHEADS * SEQ, 256, 0, stream>>>(attn);

  // 6) attn @ V  (WMMA, 16x64 per wave, NT last-use loads of attn)
  av_kernel<<<dim3((SEQ / 16) / 4, 1, BHEADS), wblk, 0, stream>>>(attn, Vh, OutH);

  // 7) output projection + bias  (WMMA, 64x16 per wave)
  proj_kernel<<<dim3(ROWS / 64, (DM / 16) / 4), wblk, 0, stream>>>(
      OutH, WoT, bo, out);
}
